// M4Fuse_44195213476102
// MI455X (gfx1250) — matrix operand order repaired and checked by reading, not compiled, not run on hardware
//
#include <hip/hip_runtime.h>
#include <hip/hip_bf16.h>
#include <math.h>

// ---------------- types / constants ----------------
typedef __attribute__((ext_vector_type(16))) __bf16  v16bf;
typedef __attribute__((ext_vector_type(8)))  float   v8f;
typedef __attribute__((ext_vector_type(4)))  unsigned int u32x4;
typedef __attribute__((ext_vector_type(2)))  unsigned int u32x2;
typedef __attribute__((ext_vector_type(4)))  float   f32x4;

#define EPS_GN 1e-5f
#define LDA_T  40   // LDS A-tile row stride in bf16 (80B rows: 16B aligned, conflict-poor)

static inline int  gdiv(long long a, long long b){ return (int)((a + b - 1) / b); }
static inline int  rup (int a, int b){ return (a + b - 1) / b * b; }

// ---------------- device helpers ----------------
__device__ __forceinline__ unsigned bfr(float f){            // f32 -> bf16 bits (RNE), in low 16
  unsigned u = __float_as_uint(f);
  return (u + 0x7fffu + ((u >> 16) & 1u)) >> 16;
}
__device__ __forceinline__ __bf16 f2bf(float f){
  unsigned short h = (unsigned short)bfr(f);
  __bf16 b; __builtin_memcpy(&b, &h, 2); return b;
}
__device__ __forceinline__ float sigmoidf_(float v){ return 1.f / (1.f + __expf(-v)); }
__device__ __forceinline__ float act_fn(float v, int act){
  if (act == 1) return 0.5f * v * (1.f + erff(v * 0.70710678118f)); // exact gelu
  if (act == 2) return v * sigmoidf_(v);                           // silu
  if (act == 3) return sigmoidf_(v);                               // sigmoid
  if (act == 4) return (v > 20.f) ? v : log1pf(__expf(v));         // softplus
  return v;
}

// fragment loaders: A layout (16-bit 16x32): lanes<16 hold K {0..7,16..23}, lanes>=16 K {8..15,24..31}
__device__ __forceinline__ v16bf frag_split(const __bf16* p){ // reads [p,p+8) and [p+16,p+24)
  u32x4 lo = *(const u32x4*)p;
  u32x4 hi = *(const u32x4*)(p + 16);
  v16bf v; __builtin_memcpy(&v, &lo, 16); __builtin_memcpy(((char*)&v) + 16, &hi, 16);
  return v;
}
__device__ __forceinline__ v16bf frag_contig(const __bf16* p){ // 16 consecutive bf16
  u32x4 lo = *(const u32x4*)p;
  u32x4 hi = *(const u32x4*)(p + 8);
  v16bf v; __builtin_memcpy(&v, &lo, 16); __builtin_memcpy(((char*)&v) + 16, &hi, 16);
  return v;
}

// 32x32 wave tile: 2 A-frags x 2 B-frags -> 4 x v_wmma_f32_16x16x32_bf16
__device__ __forceinline__ void mma_32x32(const __bf16* sA, const __bf16* __restrict__ Wp,
                                          int Kp, int n_base, int k0, int wm, int lane,
                                          v8f acc[2][2]){
  int arow = wm * 32 + (lane & 15);
  const __bf16* ap0 = sA + arow * LDA_T + ((lane & 16) ? 8 : 0);
  const __bf16* ap1 = ap0 + 16 * LDA_T;
  v16bf a0 = frag_split(ap0);
  v16bf a1 = frag_split(ap1);
  int ncol = n_base + (lane & 15);
  int kb   = k0 + ((lane & 16) ? 8 : 0);
  const __bf16* bp0 = Wp + (long long)ncol * Kp + kb;
  v16bf b0 = frag_split(bp0);
  v16bf b1 = frag_split(bp0 + (long long)16 * Kp);
  acc[0][0] = __builtin_amdgcn_wmma_f32_16x16x32_bf16(false, a0, false, b0, (short)0, acc[0][0], false, false);
  acc[0][1] = __builtin_amdgcn_wmma_f32_16x16x32_bf16(false, a0, false, b1, (short)0, acc[0][1], false, false);
  acc[1][0] = __builtin_amdgcn_wmma_f32_16x16x32_bf16(false, a1, false, b0, (short)0, acc[1][0], false, false);
  acc[1][1] = __builtin_amdgcn_wmma_f32_16x16x32_bf16(false, a1, false, b1, (short)0, acc[1][1], false, false);
}

__device__ __forceinline__ void gemm_epilogue(v8f acc[2][2], float* __restrict__ C,
                                              const float* __restrict__ bias,
                                              int M, int N, int m0, int n0,
                                              int wm, int wn, int lane, int act){
  for (int i = 0; i < 2; ++i){
    int mbase = m0 + wm * 32 + i * 16 + ((lane & 16) ? 8 : 0);
    for (int j = 0; j < 2; ++j){
      int nn = n0 + wn * 32 + j * 16 + (lane & 15);
      if (nn >= N) continue;
      float bv = bias ? bias[nn] : 0.f;
#pragma unroll
      for (int r = 0; r < 8; ++r){
        int m = mbase + r;
        if (m < M) C[(long long)m * N + nn] = act_fn(acc[i][j][r] + bv, act);
      }
    }
  }
}

// vectorized f32->bf16 staging of one 4-element group into LDS (8B aligned store)
__device__ __forceinline__ void stage_group(__bf16* sA, int r, int cg, f32x4 v){
  u32x2 p;
  p.x = bfr(v.x) | (bfr(v.y) << 16);
  p.y = bfr(v.z) | (bfr(v.w) << 16);
  *(u32x2*)(sA + r * LDA_T + cg) = p;
}

// ---------------- WMMA GEMM: C[M,N] = act(A[M,K] * Wp[N,Kp]^T + bias) ----------------
__launch_bounds__(256)
__global__ void gemm_bf16_kernel(const float* __restrict__ A, int lda,
                                 const __bf16* __restrict__ Wp, const float* __restrict__ bias,
                                 float* __restrict__ C, int M, int N, int K, int Kp, int act){
  __shared__ __bf16 sA[128 * LDA_T];
  int tid = threadIdx.x, lane = tid & 31, wave = tid >> 5;
  int wm = wave & 3, wn = wave >> 2;      // 4 x 2 waves -> 128 x 64 tile
  int m0 = blockIdx.x * 128, n0 = blockIdx.y * 64;
  v8f acc[2][2] = {};
  for (int k0 = 0; k0 < Kp; k0 += 32){
    // prefetch next K-slab of A into cache (gfx1250 global_prefetch_b8)
    if (k0 + 32 < Kp && tid < 128 && m0 + tid < M)
      __builtin_prefetch(&A[(long long)(m0 + tid) * lda + k0 + 32], 0, 3);
    // stage 128x32 A tile: float4 loads -> packed bf16x4 ds_store_b64
#pragma unroll
    for (int i = 0; i < 4; ++i){
      int e4 = tid + 256 * i;                  // 1024 groups of 4
      int r = e4 >> 3, cg = (e4 & 7) << 2;
      int m = m0 + r, k = k0 + cg;
      f32x4 v = {0.f, 0.f, 0.f, 0.f};
      if (m < M){
        const float* ap = A + (long long)m * lda + k;
        if (k + 3 < K) v = *(const f32x4*)ap;          // rows are 16B aligned (K % 4 == 0 here)
        else { for (int j = 0; j < 4; ++j) if (k + j < K) v[j] = ap[j]; }
      }
      stage_group(sA, r, cg, v);
    }
    __syncthreads();
    mma_32x32(sA, Wp, Kp, n0 + wn * 32, k0, wm, lane, acc);
    __syncthreads();
  }
  gemm_epilogue(acc, C, bias, M, N, m0, n0, wm, wn, lane, act);
}

// ------ implicit-GEMM 3x3x3 conv, channels-last x[B,D,H,W,Cin] -> y[M,Cout] ------
__launch_bounds__(256)
__global__ void conv3d_bf16_kernel(const float* __restrict__ X, const __bf16* __restrict__ Wp,
                                   const float* __restrict__ bias, float* __restrict__ C,
                                   int B, int D, int H, int W, int Cin, int Kp, int Cout, int Np){
  __shared__ __bf16 sA[128 * LDA_T];
  __shared__ long long rOff[128];
  int tid = threadIdx.x, lane = tid & 31, wave = tid >> 5;
  int wm = wave & 3, wn = wave >> 2;
  int m0 = blockIdx.x * 128, n0 = blockIdx.y * 64;
  int M = B * D * H * W;
  v8f acc[2][2] = {};
  for (int t = 0; t < 27; ++t){
    int dz = t / 9 - 1, dy = (t / 3) % 3 - 1, dx = t % 3 - 1;
    if (tid < 128){
      int v = m0 + tid; long long off = -1;
      if (v < M){
        int x = v % W, y = (v / W) % H, z = (v / (W * H)) % D, b = v / (W * H * D);
        int zz = z + dz, yy = y + dy, xx = x + dx;
        if (zz >= 0 && zz < D && yy >= 0 && yy < H && xx >= 0 && xx < W)
          off = ((long long)((b * D + zz) * H + yy) * W + xx) * Cin;
      }
      rOff[tid] = off;
    }
    __syncthreads();
    const __bf16* Wt = Wp + (long long)t * Np * Kp;
    for (int k0 = 0; k0 < Kp; k0 += 32){
#pragma unroll
      for (int i = 0; i < 4; ++i){
        int e4 = tid + 256 * i;
        int r = e4 >> 3, cg = (e4 & 7) << 2;
        long long off = rOff[r]; int k = k0 + cg;
        f32x4 v = {0.f, 0.f, 0.f, 0.f};
        if (off >= 0){
          const float* ap = X + off + k;
          if (k + 3 < Cin) v = *(const f32x4*)ap;       // voxel rows 16B aligned (Cin % 4 == 0)
          else { for (int j = 0; j < 4; ++j) if (k + j < Cin) v[j] = ap[j]; }
        }
        stage_group(sA, r, cg, v);
      }
      __syncthreads();
      mma_32x32(sA, Wt, Kp, n0 + wn * 32, k0, wm, lane, acc);
      __syncthreads();
    }
  }
  gemm_epilogue(acc, C, bias, M, Cout, m0, n0, wm, wn, lane, 0);
}

// ---------------- weight packing ----------------
__global__ void k_pack_linear(const float* __restrict__ W, __bf16* __restrict__ P,
                              int N, int K, int Kp, long long tot){
  long long i = (long long)blockIdx.x * 256 + threadIdx.x; if (i >= tot) return;
  long long n = i / Kp, k = i % Kp;
  float v = (n < N && k < K) ? W[n * K + k] : 0.f;
  P[i] = f2bf(v);
}
__global__ void k_pack_conv(const float* __restrict__ W, __bf16* __restrict__ P,
                            int Cout, int Cin, int Kp, int Np, long long tot){
  long long i = (long long)blockIdx.x * 256 + threadIdx.x; if (i >= tot) return;
  int k = (int)(i % Kp); long long r = i / Kp; int n = (int)(r % Np); int t = (int)(r / Np);
  float v = (n < Cout && k < Cin) ? W[((long long)n * Cin + k) * 27 + t] : 0.f;
  P[i] = f2bf(v);
}

// ---------------- layout transposes ----------------
__global__ void k_nchw2nlc(const float* __restrict__ X, float* __restrict__ Y,
                           int Cc, long long S, long long tot){
  long long i = (long long)blockIdx.x * 256 + threadIdx.x; if (i >= tot) return;
  int c = (int)(i % Cc); long long s = (i / Cc) % S, b = i / (Cc * S);
  Y[i] = X[(b * Cc + c) * S + s];
}
__global__ void k_nlc2nchw(const float* __restrict__ X, float* __restrict__ Y,
                           int Cc, long long S, long long tot){
  long long i = (long long)blockIdx.x * 256 + threadIdx.x; if (i >= tot) return;
  long long s = i % S; int c = (int)((i / S) % Cc); long long b = i / (S * Cc);
  Y[i] = X[(b * S + s) * Cc + c];
}

// ---------------- group norm ----------------
__global__ void k_gn_stats(const float* __restrict__ X, float* __restrict__ st,
                           int S, int Cc, int G){
  int bg = blockIdx.x, b = bg / G, g = bg % G, cg = Cc / G;
  float s = 0.f, q = 0.f;
  for (int ss = threadIdx.x; ss < S; ss += 256){
    const float* p = X + ((long long)b * S + ss) * Cc + g * cg;
    for (int c = 0; c < cg; ++c){ float v = p[c]; s += v; q += v * v; }
  }
  __shared__ float rs[256], rq[256];
  rs[threadIdx.x] = s; rq[threadIdx.x] = q; __syncthreads();
  for (int o = 128; o; o >>= 1){
    if (threadIdx.x < o){ rs[threadIdx.x] += rs[threadIdx.x + o]; rq[threadIdx.x] += rq[threadIdx.x + o]; }
    __syncthreads();
  }
  if (threadIdx.x == 0){
    float n = (float)S * cg, mu = rs[0] / n, var = rq[0] / n - mu * mu;
    st[bg * 2] = mu; st[bg * 2 + 1] = rsqrtf(var + EPS_GN);
  }
}
__global__ void k_gn_apply(const float* __restrict__ X, const float* __restrict__ st,
                           const float* __restrict__ w, const float* __restrict__ bb,
                           const float* __restrict__ skip, float* __restrict__ Y,
                           int S, int Cc, int G, int act, long long tot){
  long long i = (long long)blockIdx.x * 256 + threadIdx.x; if (i >= tot) return;
  int c = (int)(i % Cc); long long b = i / ((long long)S * Cc);
  int g = c / (Cc / G); long long bg = b * G + g;
  float v = (X[i] - st[bg * 2]) * st[bg * 2 + 1] * w[c] + bb[c];
  v = act_fn(v, act);
  if (skip) v += skip[i];
  Y[i] = v;
}

// ---------------- layer norm: one wave per row (wave32) ----------------
__global__ void k_layer_norm(const float* __restrict__ X, const float* __restrict__ w,
                             const float* __restrict__ b, float* __restrict__ Y,
                             long long rows, int Cc){
  int lane = threadIdx.x & 31, wv = threadIdx.x >> 5;
  long long row = (long long)blockIdx.x * 8 + wv;
  if (row >= rows) return;
  const float* xr = X + row * Cc;
  float buf[8]; int cnt = Cc / 32;
  float s = 0.f, q = 0.f;
  for (int i = 0; i < cnt; ++i){ float v = xr[lane + 32 * i]; buf[i] = v; s += v; q += v * v; }
  for (int o = 16; o; o >>= 1){ s += __shfl_xor(s, o); q += __shfl_xor(q, o); }
  float mu = s / Cc, rstd = rsqrtf(q / Cc - mu * mu + EPS_GN);
  float* yr = Y + row * Cc;
  for (int i = 0; i < cnt; ++i){ int c = lane + 32 * i; yr[c] = (buf[i] - mu) * rstd * w[c] + b[c]; }
}

// ---------------- mixer plumbing ----------------
__global__ void k_split4(const float* __restrict__ X, float* __restrict__ U,
                         int B, int L, int dm, int Cc, long long tot){
  long long i = (long long)blockIdx.x * 256 + threadIdx.x; if (i >= tot) return;
  int d = (int)(i % dm); long long l = (i / dm) % L, qb = i / ((long long)dm * L);
  int q = (int)(qb / B), b = (int)(qb % B);
  U[i] = X[((long long)b * L + l) * Cc + q * dm + d];
}
__global__ void k_merge4(const float* __restrict__ MO, const float* __restrict__ U,
                         const float* __restrict__ skip, float* __restrict__ XM,
                         int B, int L, int dm, int Cc, long long tot){
  long long i = (long long)blockIdx.x * 256 + threadIdx.x; if (i >= tot) return;
  int d = (int)(i % dm); long long l = (i / dm) % L, qb = i / ((long long)dm * L);
  int q = (int)(qb / B), b = (int)(qb % B);
  XM[((long long)b * L + l) * Cc + q * dm + d] = MO[i] + skip[0] * U[i];
}
__global__ void k_dwconv_silu(const float* __restrict__ XZ, int ldxz,
                              const float* __restrict__ w, const float* __restrict__ cb,
                              float* __restrict__ Y, int L, int di, long long tot){
  long long i = (long long)blockIdx.x * 256 + threadIdx.x; if (i >= tot) return;
  int d = (int)(i % di); long long m = i / di; int l = (int)(m % L);
  float acc = cb[d];
#pragma unroll
  for (int j = 0; j < 4; ++j){
    if (l + j >= 3) acc += w[d * 4 + j] * XZ[(m - 3 + j) * ldxz + d];
  }
  Y[i] = acc * sigmoidf_(acc);
}
__global__ void k_dtproj(const float* __restrict__ DBL, int ldbl, int dr,
                         const float* __restrict__ Wd, const float* __restrict__ bd,
                         float* __restrict__ DT, int di, long long tot){
  long long i = (long long)blockIdx.x * 256 + threadIdx.x; if (i >= tot) return;
  int d = (int)(i % di); long long m = i / di;
  float s = bd[d];
  for (int k = 0; k < dr; ++k) s += DBL[m * ldbl + k] * Wd[d * dr + k];
  DT[i] = act_fn(s, 4);
}
// selective scan; block per batch-slice, thread per channel d (blockDim = di <= 128)
__global__ void k_scan(const float* __restrict__ DT, const float* __restrict__ XC,
                       const float* __restrict__ DBL, int ldbl, int dr,
                       const float* __restrict__ XZ, int ldxz, int di,
                       const float* __restrict__ A_log, const float* __restrict__ Dv,
                       float* __restrict__ Y, int L){
  int b = blockIdx.x, d = threadIdx.x;
  float A[16], h[16];
#pragma unroll
  for (int n = 0; n < 16; ++n){ A[n] = -__expf(A_log[d * 16 + n]); h[n] = 0.f; }
  float Dd = Dv[d];
  __shared__ float sB[16], sC[16];
  for (int l = 0; l < L; ++l){
    long long row = (long long)b * L + l;
    if (d < 16){ sB[d] = DBL[row * ldbl + dr + d]; sC[d] = DBL[row * ldbl + dr + 16 + d]; }
    __syncthreads();
    float dtv = DT[row * di + d], xv = XC[row * di + d], dtx = dtv * xv, y = 0.f;
#pragma unroll
    for (int n = 0; n < 16; ++n){
      h[n] = __expf(dtv * A[n]) * h[n] + dtx * sB[n];
      y += h[n] * sC[n];
    }
    float zv = XZ[row * ldxz + di + d];
    Y[row * di + d] = (y + xv * Dd) * (zv * sigmoidf_(zv));
    __syncthreads();
  }
}
__global__ void k_combine3(const float* __restrict__ S, const float* __restrict__ E0,
                           const float* __restrict__ E1, const int* __restrict__ did,
                           float* __restrict__ O, long long per, int act, long long tot){
  long long i = (long long)blockIdx.x * 256 + threadIdx.x; if (i >= tot) return;
  int b = (int)(i / per);
  float v = S[i] + ((did[b] == 0) ? E0[i] : E1[i]);
  O[i] = act_fn(v, act);
}

// ---------------- pool / resize ----------------
__global__ void k_maxpool(const float* __restrict__ X, float* __restrict__ Y,
                          int D, int H, int W, int Cc, int Do, int Ho, int Wo, long long tot){
  long long i = (long long)blockIdx.x * 256 + threadIdx.x; if (i >= tot) return;
  int c = (int)(i % Cc); long long t = i / Cc;
  int xo = (int)(t % Wo), yo = (int)((t / Wo) % Ho), zo = (int)((t / ((long long)Wo * Ho)) % Do);
  int b = (int)(t / ((long long)Wo * Ho * Do));
  float mx = -3.4e38f;
#pragma unroll
  for (int dz = 0; dz < 2; ++dz)
#pragma unroll
    for (int dy = 0; dy < 2; ++dy)
#pragma unroll
      for (int dx = 0; dx < 2; ++dx){
        long long idx = ((long long)((b * D + 2 * zo + dz) * H + 2 * yo + dy) * W + 2 * xo + dx) * Cc + c;
        mx = fmaxf(mx, X[idx]);
      }
  Y[i] = mx;
}
__global__ void k_resize(const float* __restrict__ X, float* __restrict__ Y, int B,
                         int d0, int h0, int w0, int d1, int h1, int w1, int Cc, long long tot){
  long long i = (long long)blockIdx.x * 256 + threadIdx.x; if (i >= tot) return;
  int c = (int)(i % Cc); long long t = i / Cc;
  int x1 = (int)(t % w1), y1 = (int)((t / w1) % h1), z1 = (int)((t / ((long long)w1 * h1)) % d1);
  int b = (int)(t / ((long long)w1 * h1 * d1));
  float pz = z1 * (float)(d0 - 1) / (float)(d1 - 1);
  float py = y1 * (float)(h0 - 1) / (float)(h1 - 1);
  float px = x1 * (float)(w0 - 1) / (float)(w1 - 1);
  int zi = min((int)pz, d0 - 2), yi = min((int)py, h0 - 2), xi = min((int)px, w0 - 2);
  float wz = pz - zi, wy = py - yi, wx = px - xi;
  float acc = 0.f;
#pragma unroll
  for (int a = 0; a < 2; ++a)
#pragma unroll
    for (int e = 0; e < 2; ++e)
#pragma unroll
      for (int f = 0; f < 2; ++f){
        float wgt = (a ? wz : 1.f - wz) * (e ? wy : 1.f - wy) * (f ? wx : 1.f - wx);
        long long idx = ((long long)((b * d0 + zi + a) * h0 + yi + e) * w0 + xi + f) * Cc + c;
        acc += wgt * X[idx];
      }
  Y[i] = acc;
}

// ---------------- full-bloom bridge ----------------
__global__ void k_reduce_mm(const float* __restrict__ T, float* __restrict__ MM, int Cc, long long tot){
  long long i = (long long)blockIdx.x * 256 + threadIdx.x; if (i >= tot) return;
  const float* p = T + i * Cc;
  float s = 0.f, mx = -3.4e38f;
  for (int c = 0; c < Cc; ++c){ float v = p[c]; s += v; mx = fmaxf(mx, v); }
  MM[i * 2] = s / Cc; MM[i * 2 + 1] = mx;
}
__global__ void k_sconv7(const float* __restrict__ MM, const float* __restrict__ w,
                         float* __restrict__ SP, int D, int H, int W, long long tot){
  long long i = (long long)blockIdx.x * 256 + threadIdx.x; if (i >= tot) return;
  long long S = (long long)D * H * W;
  int s = (int)(i % S), b = (int)(i / S);
  int x = s % W, y = (s / W) % H, z = s / (W * H);
  float acc = 0.f;
  for (int dz = -3; dz <= 3; ++dz){ int zz = z + dz; if (zz < 0 || zz >= D) continue;
    for (int dy = -3; dy <= 3; ++dy){ int yy = y + dy; if (yy < 0 || yy >= H) continue;
      for (int dx = -3; dx <= 3; ++dx){ int xx = x + dx; if (xx < 0 || xx >= W) continue;
        long long idx = ((long long)(b * D + zz) * H + yy) * W + xx;
        int tap = ((dz + 3) * 7 + (dy + 3)) * 7 + (dx + 3);
        acc += w[tap] * MM[idx * 2] + w[343 + tap] * MM[idx * 2 + 1];
      }}}
  SP[i] = sigmoidf_(acc);
}
__global__ void k_spmul(const float* __restrict__ T, const float* __restrict__ SP,
                        float* __restrict__ Y, int Cc, long long tot){
  long long i = (long long)blockIdx.x * 256 + threadIdx.x; if (i >= tot) return;
  Y[i] = T[i] * SP[i / Cc];
}
__global__ void k_chan_mean(const float* __restrict__ TSP, float* __restrict__ pooled,
                            int S, int Cc, int chOff){
  int b = blockIdx.x / Cc, c = blockIdx.x % Cc;
  float s = 0.f;
  for (int ss = threadIdx.x; ss < S; ss += 256) s += TSP[((long long)b * S + ss) * Cc + c];
  __shared__ float red[256];
  red[threadIdx.x] = s; __syncthreads();
  for (int o = 128; o; o >>= 1){ if (threadIdx.x < o) red[threadIdx.x] += red[threadIdx.x + o]; __syncthreads(); }
  if (threadIdx.x == 0) pooled[b * 512 + chOff + c] = red[0] / S;
}
__global__ void k_catt(const float* __restrict__ pooled, const float* __restrict__ w,
                       float* __restrict__ att, long long tot){
  long long i = (long long)blockIdx.x * 256 + threadIdx.x; if (i >= tot) return;
  int b = (int)(i / 512), j = (int)(i % 512);
  float a = 0.f;
#pragma unroll
  for (int u = 0; u < 3; ++u){ int jj = j + u - 1; if (jj >= 0 && jj < 512) a += w[u] * pooled[b * 512 + jj]; }
  att[i] = a;
}
__global__ void k_fbb_att(const float* __restrict__ att, const float* __restrict__ w,
                          const float* __restrict__ bias, float* __restrict__ A,
                          int Cc, long long tot){
  long long i = (long long)blockIdx.x * 256 + threadIdx.x; if (i >= tot) return;
  int co = (int)(i % Cc), b = (int)(i / Cc);
  float s = bias[co];
  for (int j = 0; j < 512; ++j) s += att[b * 512 + j] * w[(long long)co * 512 + j];
  A[i] = sigmoidf_(s);
}
__global__ void k_fbb_out(const float* __restrict__ TSP, const float* __restrict__ A,
                          const float* __restrict__ TORIG, float* __restrict__ O,
                          int S, int Cc, long long tot){
  long long i = (long long)blockIdx.x * 256 + threadIdx.x; if (i >= tot) return;
  int c = (int)(i % Cc); long long b = i / ((long long)S * Cc);
  O[i] = TSP[i] * (1.f + A[b * Cc + c]) + TORIG[i];
}

// ============================= HOST =============================
struct Bump {
  char* base; size_t off, cap;
  void* raw(size_t bytes){ size_t o = (off + 255) & ~(size_t)255; off = o + bytes; return base + o; }
  float*  f (size_t n){ return (float*) raw(n * 4); }
  __bf16* bf(size_t n){ return (__bf16*)raw(n * 2); }
};

struct MixerP {
  const float *nw,*nb,*in_w,*conv_w,*conv_b,*xproj_w,*dtproj_w,*dtproj_b,*A_log,*Dvec,*out_w,*proj_w,*proj_b,*skip;
  int din, dout;
};
static MixerP mixer_at(void* const* din_, int base, int dinn, int dout){
  MixerP m;
  const float** f = (const float**)&m;
  for (int i = 0; i < 14; ++i) f[i] = (const float*)din_[base + i];
  m.din = dinn; m.dout = dout; return m;
}

static __bf16* packW(hipStream_t st, Bump& ws, const float* W, int N, int K, int& Kp){
  Kp = rup(K, 32); int Np = rup(N, 64);
  __bf16* P = ws.bf((size_t)Np * Kp);
  long long tot = (long long)Np * Kp;
  k_pack_linear<<<gdiv(tot, 256), 256, 0, st>>>(W, P, N, K, Kp, tot);
  return P;
}
static void gemm(hipStream_t st, const float* A, int lda, const __bf16* P, int Kp,
                 const float* bias, float* C, int M, int N, int K, int act){
  dim3 g(gdiv(M, 128), gdiv(N, 64));
  gemm_bf16_kernel<<<g, dim3(256), 0, st>>>(A, lda, P, bias, C, M, N, K, Kp, act);
}
static float* conv3(hipStream_t st, Bump& ws, const float* X, int B, int D, int H, int W,
                    int Cin, const float* Wt, const float* bias, int Cout){
  int Kp = rup(Cin, 32), Np = rup(Cout, 64);
  __bf16* P = ws.bf((size_t)27 * Np * Kp);
  long long tot = (long long)27 * Np * Kp;
  k_pack_conv<<<gdiv(tot, 256), 256, 0, st>>>(Wt, P, Cout, Cin, Kp, Np, tot);
  int M = B * D * H * W;
  float* Y = ws.f((size_t)M * Cout);
  dim3 g(gdiv(M, 128), gdiv(Cout, 64));
  conv3d_bf16_kernel<<<g, dim3(256), 0, st>>>(X, P, bias, Y, B, D, H, W, Cin, Kp, Cout, Np);
  return Y;
}
static float* gn_act(hipStream_t st, Bump& ws, const float* X, int B, int S, int C,
                     const float* w, const float* bb, int act, const float* skip){
  const int G = 4;
  float* stt = ws.f((size_t)B * G * 2);
  k_gn_stats<<<B * G, 256, 0, st>>>(X, stt, S, C, G);
  long long tot = (long long)B * S * C;
  float* Y = ws.f((size_t)tot);
  k_gn_apply<<<gdiv(tot, 256), 256, 0, st>>>(X, stt, w, bb, skip, Y, S, C, G, act, tot);
  return Y;
}
static float* pool(hipStream_t st, Bump& ws, const float* X, int B, int D, int C){
  int Do = D / 2;
  long long tot = (long long)B * Do * Do * Do * C;
  float* Y = ws.f((size_t)tot);
  k_maxpool<<<gdiv(tot, 256), 256, 0, st>>>(X, Y, D, D, D, C, Do, Do, Do, tot);
  return Y;
}
static float* resize(hipStream_t st, Bump& ws, const float* X, int B, int d0, int d1, int C){
  long long tot = (long long)B * d1 * d1 * d1 * C;
  float* Y = ws.f((size_t)tot);
  k_resize<<<gdiv(tot, 256), 256, 0, st>>>(X, Y, B, d0, d0, d0, d1, d1, d1, C, tot);
  return Y;
}

static float* run_mixer(hipStream_t st, Bump& ws, const MixerP& mp, const float* xf, int B, int L){
  const int C = mp.din, dm = C / 4, di = 2 * dm, dr = (dm + 15) / 16, B4 = 4 * B;
  const int M = B4 * L;
  float* xn = ws.f((size_t)B * L * C);
  k_layer_norm<<<gdiv((long long)B * L, 8), 256, 0, st>>>(xf, mp.nw, mp.nb, xn, (long long)B * L, C);
  float* u = ws.f((size_t)M * dm);
  { long long tot = (long long)M * dm; k_split4<<<gdiv(tot, 256), 256, 0, st>>>(xn, u, B, L, dm, C, tot); }
  int KpIn; __bf16* Pin = packW(st, ws, mp.in_w, 2 * di, dm, KpIn);
  float* xz = ws.f((size_t)M * 2 * di);
  gemm(st, u, dm, Pin, KpIn, nullptr, xz, M, 2 * di, dm, 0);
  float* xc = ws.f((size_t)M * di);
  { long long tot = (long long)M * di;
    k_dwconv_silu<<<gdiv(tot, 256), 256, 0, st>>>(xz, 2 * di, mp.conv_w, mp.conv_b, xc, L, di, tot); }
  int Nx = dr + 32; int KpX; __bf16* Px = packW(st, ws, mp.xproj_w, Nx, di, KpX);
  float* dbl = ws.f((size_t)M * Nx);
  gemm(st, xc, di, Px, KpX, nullptr, dbl, M, Nx, di, 0);
  float* dtb = ws.f((size_t)M * di);
  { long long tot = (long long)M * di;
    k_dtproj<<<gdiv(tot, 256), 256, 0, st>>>(dbl, Nx, dr, mp.dtproj_w, mp.dtproj_b, dtb, di, tot); }
  float* y = ws.f((size_t)M * di);
  k_scan<<<B4, di, 0, st>>>(dtb, xc, dbl, Nx, dr, xz, 2 * di, di, mp.A_log, mp.Dvec, y, L);
  int KpO; __bf16* Po = packW(st, ws, mp.out_w, dm, di, KpO);
  float* mo = ws.f((size_t)M * dm);
  gemm(st, y, di, Po, KpO, nullptr, mo, M, dm, di, 0);
  float* xm = ws.f((size_t)B * L * C);
  { long long tot = (long long)M * dm;
    k_merge4<<<gdiv(tot, 256), 256, 0, st>>>(mo, u, mp.skip, xm, B, L, dm, C, tot); }
  float* xn2 = ws.f((size_t)B * L * C);
  k_layer_norm<<<gdiv((long long)B * L, 8), 256, 0, st>>>(xm, mp.nw, mp.nb, xn2, (long long)B * L, C);
  int KpP; __bf16* Pp = packW(st, ws, mp.proj_w, mp.dout, C, KpP);
  float* outp = ws.f((size_t)B * L * mp.dout);
  gemm(st, xn2, C, Pp, KpP, mp.proj_b, outp, B * L, mp.dout, C, 0);
  return outp;
}
static float* run_expert(hipStream_t st, Bump& ws, void* const* din_, int base, int dinn, int dout,
                         const float* X, int B, int L, const int* did, int act){
  MixerP s  = mixer_at(din_, base,      dinn, dout);
  MixerP e0 = mixer_at(din_, base + 14, dinn, dout);
  MixerP e1 = mixer_at(din_, base + 28, dinn, dout);
  float* os = run_mixer(st, ws, s,  X, B, L);
  float* o0 = run_mixer(st, ws, e0, X, B, L);
  float* o1 = run_mixer(st, ws, e1, X, B, L);
  long long tot = (long long)B * L * dout;
  float* out = ws.f((size_t)tot);
  k_combine3<<<gdiv(tot, 256), 256, 0, st>>>(os, o0, o1, did, out, (long long)L * dout, act, tot);
  return out;
}

extern "C" void kernel_launch(void* const* d_in, const int* in_sizes, int n_in,
                              void* d_out, int out_size, void* d_ws, size_t ws_size,
                              hipStream_t stream){
  (void)in_sizes; (void)n_in; (void)out_size;
  Bump ws{(char*)d_ws, 0, ws_size};
  auto F = [&](int i){ return (const float*)d_in[i]; };
  const float* x  = F(0);
  const int* did  = (const int*)d_in[1];
  const int B = 2;
  const int D1 = 96, D2 = 48, D3 = 24, D4 = 12, D5 = 6;
  const int S1 = D1*D1*D1, S2 = D2*D2*D2, S3 = D3*D3*D3, S4 = D4*D4*D4, S5 = D5*D5*D5;

  // NCDHW -> channels-last
  float* xcl = ws.f((size_t)B * S1 * 4);
  { long long tot = (long long)B * S1 * 4;
    k_nchw2nlc<<<gdiv(tot, 256), 256, 0, stream>>>(x, xcl, 4, S1, tot); }

  // encoder
  float* c1 = conv3(stream, ws, xcl, B, D1, D1, D1, 4,  F(2), F(3), 32);
  float* t1 = gn_act(stream, ws, c1, B, S1, 32,  F(194), F(195), 1, nullptr);
  float* p1 = pool(stream, ws, t1, B, D1, 32);
  float* c2 = conv3(stream, ws, p1, B, D2, D2, D2, 32, F(4), F(5), 64);
  float* t2 = gn_act(stream, ws, c2, B, S2, 64,  F(196), F(197), 1, nullptr);
  float* p2 = pool(stream, ws, t2, B, D2, 64);
  float* c3 = conv3(stream, ws, p2, B, D3, D3, D3, 64, F(6), F(7), 96);
  float* t3 = gn_act(stream, ws, c3, B, S3, 96,  F(198), F(199), 1, nullptr);
  float* p3 = pool(stream, ws, t3, B, D3, 96);
  float* e4 = run_expert(stream, ws, d_in, 8,  96,  128, p3, B, S4, did, 0);
  float* t4 = gn_act(stream, ws, e4, B, S4, 128, F(200), F(201), 1, nullptr);
  float* p4 = pool(stream, ws, t4, B, D4, 128);
  float* e5 = run_expert(stream, ws, d_in, 50, 128, 192, p4, B, S5, did, 0);
  float* t5 = gn_act(stream, ws, e5, B, S5, 192, F(202), F(203), 1, nullptr);

  // full-bloom bridge
  struct Lv { const float* t; int S, C, D; } lv[5] = {
    {t1, S1, 32, D1}, {t2, S2, 64, D2}, {t3, S3, 96, D3}, {t4, S4, 128, D4}, {t5, S5, 192, D5}};
  const float* sconv = F(134); const float* catt = F(135);
  float* tsp[5]; float* fbb[5];
  for (int i = 0; i < 5; ++i){
    int S = lv[i].S, C = lv[i].C, D = lv[i].D;
    float* mm = ws.f((size_t)B * S * 2);
    { long long tot = (long long)B * S; k_reduce_mm<<<gdiv(tot, 256), 256, 0, stream>>>(lv[i].t, mm, C, tot); }
    float* sp = ws.f((size_t)B * S);
    { long long tot = (long long)B * S; k_sconv7<<<gdiv(tot, 256), 256, 0, stream>>>(mm, sconv, sp, D, D, D, tot); }
    tsp[i] = ws.f((size_t)B * S * C);
    { long long tot = (long long)B * S * C; k_spmul<<<gdiv(tot, 256), 256, 0, stream>>>(lv[i].t, sp, tsp[i], C, tot); }
  }
  float* pooled = ws.f((size_t)B * 512);
  { int off = 0;
    for (int i = 0; i < 5; ++i){
      k_chan_mean<<<B * lv[i].C, 256, 0, stream>>>(tsp[i], pooled, lv[i].S, lv[i].C, off);
      off += lv[i].C;
    } }
  float* att = ws.f((size_t)B * 512);
  { long long tot = (long long)B * 512; k_catt<<<gdiv(tot, 256), 256, 0, stream>>>(pooled, catt, att, tot); }
  for (int i = 0; i < 5; ++i){
    int S = lv[i].S, C = lv[i].C;
    float* a = ws.f((size_t)B * C);
    { long long tot = (long long)B * C;
      k_fbb_att<<<gdiv(tot, 256), 256, 0, stream>>>(att, F(136 + 2 * i), F(137 + 2 * i), a, C, tot); }
    fbb[i] = ws.f((size_t)B * S * C);
    { long long tot = (long long)B * S * C;
      k_fbb_out<<<gdiv(tot, 256), 256, 0, stream>>>(tsp[i], a, lv[i].t, fbb[i], S, C, tot); }
  }

  // decoder
  float* o6  = run_expert(stream, ws, d_in, 92, 192, 256, t5, B, S5, did, 1); // gelu fused
  float* m1  = run_mixer(stream, ws, mixer_at(d_in, 146, 256, 192), o6, B, S5);
  float* d1o = gn_act(stream, ws, m1, B, S5, 192, F(204), F(205), 1, fbb[4]);
  float* r1  = resize(stream, ws, d1o, B, D5, D4, 192);
  float* m2  = run_mixer(stream, ws, mixer_at(d_in, 160, 192, 128), r1, B, S4);
  float* d2o = gn_act(stream, ws, m2, B, S4, 128, F(206), F(207), 1, fbb[3]);
  float* r2  = resize(stream, ws, d2o, B, D4, D3, 128);
  float* m3  = run_mixer(stream, ws, mixer_at(d_in, 174, 128, 96), r2, B, S3);
  float* d3o = gn_act(stream, ws, m3, B, S3, 96, F(208), F(209), 1, fbb[2]);
  float* r3  = resize(stream, ws, d3o, B, D3, D2, 96);
  float* c4  = conv3(stream, ws, r3, B, D2, D2, D2, 96, F(188), F(189), 64);
  float* d4o = gn_act(stream, ws, c4, B, S2, 64, F(210), F(211), 1, fbb[1]);
  float* r4  = resize(stream, ws, d4o, B, D2, D1, 64);
  float* c5  = conv3(stream, ws, r4, B, D1, D1, D1, 64, F(190), F(191), 32);
  float* d5o = gn_act(stream, ws, c5, B, S1, 32, F(212), F(213), 1, fbb[0]);

  // final 1x1x1 conv as GEMM, then back to NCDHW
  int KpF; __bf16* Pf = packW(stream, ws, F(192), 4, 32, KpF);
  float* fo = ws.f((size_t)B * S1 * 4);
  gemm(stream, d5o, 32, Pf, KpF, F(193), fo, B * S1, 4, 32, 0);
  { long long tot = (long long)B * 4 * S1;
    k_nlc2nchw<<<gdiv(tot, 256), 256, 0, stream>>>(fo, (float*)d_out, 4, S1, tot); }
}
